// FocusedAttnDecoder_85246510891445
// MI455X (gfx1250) — matrix operand order repaired and checked by reading, not compiled, not run on hardware
//
#include <hip/hip_runtime.h>

// ---------------------------------------------------------------------------
// FocusedAttnDecoder on gfx1250 (MI455X): all GEMMs via v_wmma_f32_16x16x32_f16
// ---------------------------------------------------------------------------

typedef _Float16 half_t;
typedef __attribute__((ext_vector_type(16))) _Float16 v16h;
typedef __attribute__((ext_vector_type(8)))  float    v8f;

#define D_MODEL 256
#define N_HEADS 8
#define D_HEAD  32
#define N_LAYERS 6
#define D_FF 1024
#define S_TOT 15768
#define S_PAD 15776   // 32 * 493
#define NQ    300
#define Q_PAD 320     // 32 * 10

// ---------------------------------------------------------------------------
// WMMA GEMM:  C[z] = alpha * (A[z] * B[z]^T + bias) , optional ReLU
//   A : [M x K] f16 row-major, row stride lda, per-z elem offset a_zoff
//   B : [N x K] f16 row-major, row stride ldb, per-z elem offset b_zoff
//   NT : N-tiles per wave (wave computes 16 x 16*NT output tile)
//   CHT: 1 -> write Ch transposed as Ch[n*ldch + m] (f16), else Ch[m*ldch + n]
// Requirements: M % 16 == 0, N % (16*NT) == 0, K % 32 == 0, all buffers padded
// so every tile is fully in-bounds (no guards). blockDim = 32 (one wave).
// ---------------------------------------------------------------------------
template<int NT, int CHT>
__global__ void wmma_gemm_t(const half_t* __restrict__ A, int lda, long a_zoff,
                            const half_t* __restrict__ B, int ldb, long b_zoff,
                            const float* __restrict__ bias, float alpha, int relu,
                            float* __restrict__ Cf, int ldcf, long cf_zoff,
                            half_t* __restrict__ Ch, int ldch, long ch_zoff,
                            int K)
{
    const int z    = blockIdx.z;
    const int m0   = blockIdx.y * 16;
    const int n0   = blockIdx.x * (16 * NT);
    const int lane = threadIdx.x & 31;
    const int hsel = lane >> 4;    // K-half this lane holds
    const int l16  = lane & 15;

    const half_t* Az = A + (long)z * a_zoff + (long)(m0 + l16) * lda;
    const half_t* Bz = B + (long)z * b_zoff;

    v8f acc[NT];
#pragma unroll
    for (int j = 0; j < NT; ++j) acc[j] = (v8f){};

    for (int k0 = 0; k0 < K; k0 += 32) {
        // A fragment: lanes 0-15 hold K 0..15, lanes 16-31 K 16..31
        const half_t* ap = Az + k0 + 16 * hsel;
        v16h a;
#pragma unroll
        for (int e = 0; e < 8; ++e)  a[e] = ap[e];        // contiguous 16B
#pragma unroll
        for (int e = 8; e < 16; ++e) a[e] = ap[e];        // contiguous 16B
#pragma unroll
        for (int j = 0; j < NT; ++j) {
            // B fragment: col n0+16j+l16 ; lanes 0-15 K 0..15, lanes 16-31 K 16..31
            const half_t* bp = Bz + (long)(n0 + 16 * j + l16) * ldb + k0 + 16 * hsel;
            v16h b;
#pragma unroll
            for (int e = 0; e < 16; ++e) b[e] = bp[e];    // contiguous 32B
            acc[j] = __builtin_amdgcn_wmma_f32_16x16x32_f16(
                false, a, false, b, (short)0, acc[j], false, false);
        }
    }

    if (Cf) {
        float* cfz = Cf + (long)z * cf_zoff;
#pragma unroll
        for (int j = 0; j < NT; ++j) {
            int n = n0 + 16 * j + l16;
            float bv = bias ? bias[n] : 0.0f;
#pragma unroll
            for (int r = 0; r < 8; ++r) {
                int m = m0 + r + 8 * hsel;
                float v = (acc[j][r] + bv) * alpha;
                if (relu && v < 0.0f) v = 0.0f;
                cfz[(long)m * ldcf + n] = v;
            }
        }
    }
    if (Ch) {
        half_t* chz = Ch + (long)z * ch_zoff;
#pragma unroll
        for (int j = 0; j < NT; ++j) {
            int n = n0 + 16 * j + l16;
            float bv = bias ? bias[n] : 0.0f;
#pragma unroll
            for (int r = 0; r < 8; ++r) {
                int m = m0 + r + 8 * hsel;
                float v = (acc[j][r] + bv) * alpha;
                if (relu && v < 0.0f) v = 0.0f;
                if (CHT) chz[(long)n * ldch + m] = (half_t)v;
                else     chz[(long)m * ldch + n] = (half_t)v;
            }
        }
    }
}

// f32 -> f16 conversion (weights)
__global__ void cvt_f32_f16(const float* __restrict__ in, half_t* __restrict__ out, long n)
{
    long i = (long)blockIdx.x * blockDim.x + threadIdx.x;
    if (i < n) out[i] = (half_t)in[i];
}

// Transpose srcs/pos [C][S] -> packed f16 [S_PAD][C]: kv = src+pos, sr = src
__global__ void prep_src(const float* __restrict__ srcs, const float* __restrict__ pos,
                         half_t* __restrict__ kv_h, half_t* __restrict__ sr_h)
{
    int s = blockIdx.x * blockDim.x + threadIdx.x;
    int c = blockIdx.y;
    if (s >= S_PAD) return;
    half_t kv = (half_t)0.0f, sr = (half_t)0.0f;
    if (s < S_TOT) {
        float a = srcs[(long)c * S_TOT + s];
        float p = pos [(long)c * S_TOT + s];
        sr = (half_t)a;
        kv = (half_t)(a + p);
    }
    kv_h[(long)s * D_MODEL + c] = kv;
    sr_h[(long)s * D_MODEL + c] = sr;
}

// x init from query_embed[:, C:2C]; pad rows zero
__global__ void init_x(const float* __restrict__ qe, float* __restrict__ x)
{
    int i = blockIdx.x * blockDim.x + threadIdx.x;
    if (i >= Q_PAD * D_MODEL) return;
    int q = i / D_MODEL, c = i % D_MODEL;
    x[i] = (q < NQ) ? qe[(long)q * 2 * D_MODEL + D_MODEL + c] : 0.0f;
}

// Build f16 GEMM inputs: xq_h = f16(x + query_pos), x_h = f16(x)
__global__ void make_inputs(const float* __restrict__ x, const float* __restrict__ qe,
                            half_t* __restrict__ xq_h, half_t* __restrict__ x_h)
{
    int i = blockIdx.x * blockDim.x + threadIdx.x;
    if (i >= Q_PAD * D_MODEL) return;
    int q = i / D_MODEL, c = i % D_MODEL;
    float xv = x[i];
    if (x_h)  x_h[i]  = (half_t)xv;
    if (xq_h) {
        float pe = (q < NQ) ? qe[(long)q * 2 * D_MODEL + c] : 0.0f;
        xq_h[i] = (half_t)(xv + pe);
    }
}

// Row softmax over f16 scores (in-place -> f16 probs). mask may be null.
// grid = (Qp, H); blockDim = 256. Padded cols/rows get prob 0.
__global__ void softmax_rows(half_t* __restrict__ scores, const unsigned char* __restrict__ mask,
                             int S, int Sp, int Q, int Qp)
{
    int q = blockIdx.x, h = blockIdx.y, t = threadIdx.x;
    half_t* row = scores + ((long)h * Qp + q) * Sp;
    __shared__ float red[256];
    if (q >= Q) {
        for (int s = t; s < Sp; s += 256) row[s] = (half_t)0.0f;
        return;
    }
    const unsigned char* mrow = mask ? (mask + (long)q * S) : nullptr;
    float mx = -3.0e38f;
    for (int s = t; s < S; s += 256)
        if (!mrow || !mrow[s]) { float v = (float)row[s]; mx = fmaxf(mx, v); }
    red[t] = mx; __syncthreads();
    for (int o = 128; o > 0; o >>= 1) { if (t < o) red[t] = fmaxf(red[t], red[t + o]); __syncthreads(); }
    mx = red[0]; __syncthreads();
    float sum = 0.0f;
    for (int s = t; s < S; s += 256)
        if (!mrow || !mrow[s]) sum += __expf((float)row[s] - mx);
    red[t] = sum; __syncthreads();
    for (int o = 128; o > 0; o >>= 1) { if (t < o) red[t] += red[t + o]; __syncthreads(); }
    sum = red[0];
    float inv = (sum > 0.0f) ? (1.0f / sum) : 0.0f;
    for (int s = t; s < Sp; s += 256) {
        float p = 0.0f;
        if (s < S && (!mrow || !mrow[s])) p = __expf((float)row[s] - mx) * inv;
        row[s] = (half_t)p;
    }
}

// x = LayerNorm(x + y) * w + b ; one block (256 threads) per row
__global__ void add_ln(float* __restrict__ x, const float* __restrict__ y,
                       const float* __restrict__ w, const float* __restrict__ b)
{
    int q = blockIdx.x, t = threadIdx.x;
    __shared__ float red[256];
    float v = x[(long)q * D_MODEL + t] + y[(long)q * D_MODEL + t];
    red[t] = v; __syncthreads();
    for (int o = 128; o > 0; o >>= 1) { if (t < o) red[t] += red[t + o]; __syncthreads(); }
    float mean = red[0] * (1.0f / D_MODEL); __syncthreads();
    float d = v - mean;
    red[t] = d * d; __syncthreads();
    for (int o = 128; o > 0; o >>= 1) { if (t < o) red[t] += red[t + o]; __syncthreads(); }
    float var = red[0] * (1.0f / D_MODEL);
    x[(long)q * D_MODEL + t] = d * rsqrtf(var + 1e-5f) * w[t] + b[t];
}

__global__ void copy_out(const float* __restrict__ x, float* __restrict__ out)
{
    int i = blockIdx.x * blockDim.x + threadIdx.x;
    if (i < NQ * D_MODEL) out[i] = x[i];
}

// ---------------------------------------------------------------------------
// Host-side orchestration
// ---------------------------------------------------------------------------
static void gemm(hipStream_t st,
                 const half_t* A, int lda, long az,
                 const half_t* B, int ldb, long bz,
                 const float* bias, float alpha, int relu,
                 float* Cf, int ldcf, long cfz,
                 half_t* Ch, int ldch, long chz, int chT,
                 int M, int N, int K, int Z)
{
#define LNCH(NT, CHT) do {                                                    \
        dim3 g(N / (16 * (NT)), M / 16, Z);                                   \
        hipLaunchKernelGGL((wmma_gemm_t<NT, CHT>), g, dim3(32), 0, st,        \
                           A, lda, az, B, ldb, bz, bias, alpha, relu,         \
                           Cf, ldcf, cfz, Ch, ldch, chz, K);                  \
    } while (0)
    if (chT) {
        if      (N % 64 == 0) LNCH(4, 1);
        else if (N % 32 == 0) LNCH(2, 1);
        else                  LNCH(1, 1);
    } else {
        if      (N % 64 == 0) LNCH(4, 0);
        else if (N % 32 == 0) LNCH(2, 0);
        else                  LNCH(1, 0);
    }
#undef LNCH
}

static inline void cvt(hipStream_t st, const float* in, half_t* out, long n)
{
    int blocks = (int)((n + 255) / 256);
    hipLaunchKernelGGL(cvt_f32_f16, dim3(blocks), dim3(256), 0, st, in, out, n);
}

extern "C" void kernel_launch(void* const* d_in, const int* in_sizes, int n_in,
                              void* d_out, int out_size, void* d_ws, size_t ws_size,
                              hipStream_t stream)
{
    const float* srcs      = (const float*)d_in[0];
    const float* pos       = (const float*)d_in[1];
    const float* query_emb = (const float*)d_in[2];
    const unsigned char* attn_mask = (const unsigned char*)d_in[3];
    const float* sa_in_w  = (const float*)d_in[4];
    const float* sa_in_b  = (const float*)d_in[5];
    const float* sa_out_w = (const float*)d_in[6];
    const float* sa_out_b = (const float*)d_in[7];
    const float* ca_in_w  = (const float*)d_in[8];
    const float* ca_in_b  = (const float*)d_in[9];
    const float* ca_out_w = (const float*)d_in[10];
    const float* ca_out_b = (const float*)d_in[11];
    const float* ln1_w = (const float*)d_in[12];
    const float* ln1_b = (const float*)d_in[13];
    const float* ln2_w = (const float*)d_in[14];
    const float* ln2_b = (const float*)d_in[15];
    const float* ln3_w = (const float*)d_in[16];
    const float* ln3_b = (const float*)d_in[17];
    const float* ff1_w = (const float*)d_in[18];
    const float* ff1_b = (const float*)d_in[19];
    const float* ff2_w = (const float*)d_in[20];
    const float* ff2_b = (const float*)d_in[21];

    // ---- workspace arena --------------------------------------------------
    char* p = (char*)d_ws;
    auto alloc = [&](size_t bytes) -> void* {
        void* r = (void*)p;
        p += (bytes + 255) & ~(size_t)255;
        return r;
    };
    const long n_sa_in  = (long)N_LAYERS * 3 * D_MODEL * D_MODEL;
    const long n_sa_out = (long)N_LAYERS * D_MODEL * D_MODEL;
    const long n_ff1    = (long)N_LAYERS * D_FF * D_MODEL;
    half_t* w_sa_in_h  = (half_t*)alloc(n_sa_in  * 2);
    half_t* w_sa_out_h = (half_t*)alloc(n_sa_out * 2);
    half_t* w_ca_in_h  = (half_t*)alloc(n_sa_in  * 2);
    half_t* w_ca_out_h = (half_t*)alloc(n_sa_out * 2);
    half_t* w_ff1_h    = (half_t*)alloc(n_ff1 * 2);
    half_t* w_ff2_h    = (half_t*)alloc(n_ff1 * 2);
    half_t* kv_in_h = (half_t*)alloc((long)S_PAD * D_MODEL * 2);   // f16(src+pos), [S_PAD][C]
    half_t* src_t_h = (half_t*)alloc((long)S_PAD * D_MODEL * 2);   // f16(src),     [S_PAD][C]
    half_t* Kca_h   = (half_t*)alloc((long)S_PAD * D_MODEL * 2);   // [S_PAD][C]
    half_t* Vt_h    = (half_t*)alloc((long)D_MODEL * S_PAD * 2);   // V transposed [C][S_PAD]
    half_t* probs_ca = (half_t*)alloc((long)N_HEADS * Q_PAD * S_PAD * 2);
    half_t* probs_sa = (half_t*)alloc((long)N_HEADS * Q_PAD * Q_PAD * 2);
    half_t* q_h   = (half_t*)alloc((long)Q_PAD * D_MODEL * 2);
    half_t* k_h   = (half_t*)alloc((long)Q_PAD * D_MODEL * 2);
    half_t* vt_h  = (half_t*)alloc((long)D_MODEL * Q_PAD * 2);     // SA V transposed [C][Q_PAD]
    half_t* ctx_h = (half_t*)alloc((long)Q_PAD * D_MODEL * 2);
    half_t* xq_h  = (half_t*)alloc((long)Q_PAD * D_MODEL * 2);
    half_t* x_h   = (half_t*)alloc((long)Q_PAD * D_MODEL * 2);
    half_t* ffh_h = (half_t*)alloc((long)Q_PAD * D_FF * 2);
    float*  x     = (float*)alloc((long)Q_PAD * D_MODEL * 4);
    float*  tmp   = (float*)alloc((long)Q_PAD * D_MODEL * 4);

    const float scale = 0.17677669529663687f;   // 1/sqrt(32)
    const int QC_BLKS = (Q_PAD * D_MODEL + 255) / 256;

    // ---- prep -------------------------------------------------------------
    cvt(stream, sa_in_w,  w_sa_in_h,  n_sa_in);
    cvt(stream, sa_out_w, w_sa_out_h, n_sa_out);
    cvt(stream, ca_in_w,  w_ca_in_h,  n_sa_in);
    cvt(stream, ca_out_w, w_ca_out_h, n_sa_out);
    cvt(stream, ff1_w,    w_ff1_h,    n_ff1);
    cvt(stream, ff2_w,    w_ff2_h,    n_ff1);
    hipLaunchKernelGGL(prep_src, dim3((S_PAD + 255) / 256, D_MODEL), dim3(256), 0, stream,
                       srcs, pos, kv_in_h, src_t_h);
    hipLaunchKernelGGL(init_x, dim3(QC_BLKS), dim3(256), 0, stream, query_emb, x);

    for (int l = 0; l < N_LAYERS; ++l) {
        const half_t* wsi = w_sa_in_h  + (long)l * 3 * D_MODEL * D_MODEL;
        const half_t* wso = w_sa_out_h + (long)l * D_MODEL * D_MODEL;
        const half_t* wci = w_ca_in_h  + (long)l * 3 * D_MODEL * D_MODEL;
        const half_t* wco = w_ca_out_h + (long)l * D_MODEL * D_MODEL;
        const half_t* wf1 = w_ff1_h + (long)l * D_FF * D_MODEL;
        const half_t* wf2 = w_ff2_h + (long)l * D_MODEL * D_FF;
        const float* bsi = sa_in_b + (long)l * 3 * D_MODEL;
        const float* bso = sa_out_b + (long)l * D_MODEL;
        const float* bci = ca_in_b + (long)l * 3 * D_MODEL;
        const float* bco = ca_out_b + (long)l * D_MODEL;
        const float* bf1 = ff1_b + (long)l * D_FF;
        const float* bf2 = ff2_b + (long)l * D_MODEL;

        // ================= self-attention over queries =================
        hipLaunchKernelGGL(make_inputs, dim3(QC_BLKS), dim3(256), 0, stream, x, query_emb, xq_h, x_h);
        gemm(stream, xq_h, D_MODEL, 0, wsi, D_MODEL, 0, bsi, scale, 0,
             nullptr, 0, 0, q_h, D_MODEL, 0, 0, Q_PAD, D_MODEL, D_MODEL, 1);
        gemm(stream, xq_h, D_MODEL, 0, wsi + (long)D_MODEL * D_MODEL, D_MODEL, 0, bsi + D_MODEL, 1.0f, 0,
             nullptr, 0, 0, k_h, D_MODEL, 0, 0, Q_PAD, D_MODEL, D_MODEL, 1);
        // V stored transposed: vt_h[c][q]
        gemm(stream, x_h, D_MODEL, 0, wsi + (long)2 * D_MODEL * D_MODEL, D_MODEL, 0, bsi + 2 * D_MODEL, 1.0f, 0,
             nullptr, 0, 0, vt_h, Q_PAD, 0, 1, Q_PAD, D_MODEL, D_MODEL, 1);
        // scores[h] = q[:, h*32:] @ k[:, h*32:]^T
        gemm(stream, q_h, D_MODEL, D_HEAD, k_h, D_MODEL, D_HEAD, nullptr, 1.0f, 0,
             nullptr, 0, 0, probs_sa, Q_PAD, (long)Q_PAD * Q_PAD, 0, Q_PAD, Q_PAD, D_HEAD, N_HEADS);
        hipLaunchKernelGGL(softmax_rows, dim3(Q_PAD, N_HEADS), dim3(256), 0, stream,
                           probs_sa, (const unsigned char*)nullptr, NQ, Q_PAD, NQ, Q_PAD);
        // ctx[:, h*32:] = P[h] @ V[h]  with B = vt_h rows h*32..h*32+31 (A*B^T form)
        gemm(stream, probs_sa, Q_PAD, (long)Q_PAD * Q_PAD, vt_h, Q_PAD, (long)D_HEAD * Q_PAD,
             nullptr, 1.0f, 0, nullptr, 0, 0, ctx_h, D_MODEL, D_HEAD, 0,
             Q_PAD, D_HEAD, Q_PAD, N_HEADS);
        gemm(stream, ctx_h, D_MODEL, 0, wso, D_MODEL, 0, bso, 1.0f, 0,
             tmp, D_MODEL, 0, nullptr, 0, 0, 0, Q_PAD, D_MODEL, D_MODEL, 1);
        hipLaunchKernelGGL(add_ln, dim3(Q_PAD), dim3(D_MODEL), 0, stream,
                           x, tmp, ln2_w + (long)l * D_MODEL, ln2_b + (long)l * D_MODEL);

        // ================= masked cross-attention ======================
        hipLaunchKernelGGL(make_inputs, dim3(QC_BLKS), dim3(256), 0, stream, x, query_emb, xq_h, x_h);
        gemm(stream, xq_h, D_MODEL, 0, wci, D_MODEL, 0, bci, scale, 0,
             nullptr, 0, 0, q_h, D_MODEL, 0, 0, Q_PAD, D_MODEL, D_MODEL, 1);
        // K_ca = (src+pos) @ Wk^T + bk     [S_PAD][C]
        gemm(stream, kv_in_h, D_MODEL, 0, wci + (long)D_MODEL * D_MODEL, D_MODEL, 0, bci + D_MODEL, 1.0f, 0,
             nullptr, 0, 0, Kca_h, D_MODEL, 0, 0, S_PAD, D_MODEL, D_MODEL, 1);
        // V_ca stored transposed: Vt_h[c][s]
        gemm(stream, src_t_h, D_MODEL, 0, wci + (long)2 * D_MODEL * D_MODEL, D_MODEL, 0, bci + 2 * D_MODEL, 1.0f, 0,
             nullptr, 0, 0, Vt_h, S_PAD, 0, 1, S_PAD, D_MODEL, D_MODEL, 1);
        // scores[h] = q[:, h*32:] @ K_ca[:, h*32:]^T
        gemm(stream, q_h, D_MODEL, D_HEAD, Kca_h, D_MODEL, D_HEAD, nullptr, 1.0f, 0,
             nullptr, 0, 0, probs_ca, S_PAD, (long)Q_PAD * S_PAD, 0, Q_PAD, S_PAD, D_HEAD, N_HEADS);
        hipLaunchKernelGGL(softmax_rows, dim3(Q_PAD, N_HEADS), dim3(256), 0, stream,
                           probs_ca, attn_mask, S_TOT, S_PAD, NQ, Q_PAD);
        gemm(stream, probs_ca, S_PAD, (long)Q_PAD * S_PAD, Vt_h, S_PAD, (long)D_HEAD * S_PAD,
             nullptr, 1.0f, 0, nullptr, 0, 0, ctx_h, D_MODEL, D_HEAD, 0,
             Q_PAD, D_HEAD, S_PAD, N_HEADS);
        gemm(stream, ctx_h, D_MODEL, 0, wco, D_MODEL, 0, bco, 1.0f, 0,
             tmp, D_MODEL, 0, nullptr, 0, 0, 0, Q_PAD, D_MODEL, D_MODEL, 1);
        hipLaunchKernelGGL(add_ln, dim3(Q_PAD), dim3(D_MODEL), 0, stream,
                           x, tmp, ln1_w + (long)l * D_MODEL, ln1_b + (long)l * D_MODEL);

        // ================= FFN =========================================
        hipLaunchKernelGGL(make_inputs, dim3(QC_BLKS), dim3(256), 0, stream,
                           x, query_emb, (half_t*)nullptr, x_h);
        gemm(stream, x_h, D_MODEL, 0, wf1, D_MODEL, 0, bf1, 1.0f, /*relu=*/1,
             nullptr, 0, 0, ffh_h, D_FF, 0, 0, Q_PAD, D_FF, D_MODEL, 1);
        gemm(stream, ffh_h, D_FF, 0, wf2, D_FF, 0, bf2, 1.0f, 0,
             tmp, D_MODEL, 0, nullptr, 0, 0, 0, Q_PAD, D_MODEL, D_FF, 1);
        hipLaunchKernelGGL(add_ln, dim3(Q_PAD), dim3(D_MODEL), 0, stream,
                           x, tmp, ln3_w + (long)l * D_MODEL, ln3_b + (long)l * D_MODEL);
    }

    hipLaunchKernelGGL(copy_out, dim3((NQ * D_MODEL + 255) / 256), dim3(256), 0, stream,
                       x, (float*)d_out);
}
